// TriangleAttention_21440476741692
// MI455X (gfx1250) — matrix operand order repaired and checked by reading, not compiled, not run on hardware
//
#include <hip/hip_runtime.h>
#include <hip/hip_bf16.h>

// ---------------------------------------------------------------------------
// Triangle attention for gfx1250 (MI455X), all matmuls via v_wmma_f32_16x16x32_f16.
// S=256, D=128, H=4, HD=32.  fp32 in/out, fp16 intermediates (f32 accumulate).
// All C-tile writebacks staged through LDS -> coalesced global_store_b128.
// ---------------------------------------------------------------------------

typedef __attribute__((ext_vector_type(16))) _Float16 v16h;
typedef __attribute__((ext_vector_type(8)))  _Float16 v8h;
typedef __attribute__((ext_vector_type(4)))  _Float16 v4h;
typedef __attribute__((ext_vector_type(2)))  _Float16 v2h;
typedef __attribute__((ext_vector_type(8)))  float    v8f;

#define S_DIM   256
#define D_DIM   128
#define H_DIM   4
#define HD_DIM  32
#define NPOS    (S_DIM * S_DIM)        // 65536 positions
#define EPSLN   1e-5f
#define SCALE   0.17677669529663687f   // 32^-0.5

// ------------------------- helpers -----------------------------------------

__device__ __forceinline__ float wave_sum32(float v) {
#pragma unroll
  for (int m = 16; m >= 1; m >>= 1) v += __shfl_xor(v, m, 32);
  return v;
}

// WMMA 16-bit A-matrix half index -> K offset (contiguous: K = 16*g + t, matches B).
// lane group g = lane/16, half t = 2*v + p (v = VGPR 0..7, p = lo/hi half)
__device__ __forceinline__ int a_kmap(int g, int v) {
  return 16 * g + 2 * v;   // pair base (p in {0,1})
}

// ------------------------- kernel 1: layernorm + triangle bias --------------
// one wave per (i,j) position; 4 floats per lane.
__global__ void ln_bias_kernel(const float* __restrict__ x,
                               const float* __restrict__ ln_w,
                               const float* __restrict__ ln_b,
                               const float* __restrict__ w_bias,
                               _Float16* __restrict__ h16,
                               float* __restrict__ bias) {
  const int wave = threadIdx.x >> 5, lane = threadIdx.x & 31;
  const int pos = blockIdx.x * 8 + wave;

  const float4 xv = *(const float4*)(x + (size_t)pos * D_DIM + lane * 4);
  float s = xv.x + xv.y + xv.z + xv.w;
  s = wave_sum32(s);
  const float mu = s * (1.0f / D_DIM);

  const float d0 = xv.x - mu, d1 = xv.y - mu, d2 = xv.z - mu, d3 = xv.w - mu;
  float vs = d0 * d0 + d1 * d1 + d2 * d2 + d3 * d3;
  vs = wave_sum32(vs);
  const float rstd = rsqrtf(vs * (1.0f / D_DIM) + EPSLN);

  const float4 wv = *(const float4*)(ln_w + lane * 4);
  const float4 bv = *(const float4*)(ln_b + lane * 4);
  float hn[4];
  hn[0] = d0 * rstd * wv.x + bv.x;
  hn[1] = d1 * rstd * wv.y + bv.y;
  hn[2] = d2 * rstd * wv.z + bv.z;
  hn[3] = d3 * rstd * wv.w + bv.w;

  v4h o;
  o[0] = (_Float16)hn[0]; o[1] = (_Float16)hn[1];
  o[2] = (_Float16)hn[2]; o[3] = (_Float16)hn[3];
  *(v4h*)(h16 + (size_t)pos * D_DIM + lane * 4) = o;

#pragma unroll
  for (int hh = 0; hh < H_DIM; ++hh) {
    const float4 wb = *(const float4*)(w_bias + hh * D_DIM + lane * 4);
    float p = hn[0] * wb.x + hn[1] * wb.y + hn[2] * wb.z + hn[3] * wb.w;
    p = wave_sum32(p);
    if (lane == 0) bias[(size_t)hh * NPOS + pos] = p;
  }
}

// ------------------------- kernel 2: repack weights into B fragments --------
// B fragment (16x16x32 f16): lane -> N = lane%16, K = 16*(lane/16) + t.
// Fragment layout: wfrag[m][((j*4 + c)*32 + lane)*16 + t], m in {q,k,v,g,o}.
__global__ void prepack_kernel(const float* __restrict__ wq,
                               const float* __restrict__ wk,
                               const float* __restrict__ wv,
                               const float* __restrict__ wg,
                               const float* __restrict__ wo,
                               _Float16* __restrict__ wfrag) {
  const float* Ws[5] = {wq, wk, wv, wg, wo};
  const int m = blockIdx.x;
  const float* W = Ws[m];
  _Float16* dst = wfrag + (size_t)m * 16384;
  for (int idx = threadIdx.x; idx < 16384; idx += blockDim.x) {
    const int t = idx & 15;
    const int lane = (idx >> 4) & 31;
    const int c = (idx >> 9) & 3;
    const int j = idx >> 11;
    const int n = j * 16 + (lane & 15);
    const int k = c * 32 + (lane >> 4) * 16 + t;
    dst[idx] = (_Float16)W[n * D_DIM + k];   // y = h @ W^T  =>  B[k][n] = W[n][k]
  }
}

// ------------------------- kernel 3: q/k/v/g projections --------------------
// one wave per 16-row strip; A fragments reused across 4 outputs x 8 N-tiles.
// C tiles scattered into LDS, flushed as coalesced global_store_b128.
__global__ void proj_gemm_kernel(const _Float16* __restrict__ h16,
                                 const _Float16* __restrict__ wfrag,
                                 _Float16* __restrict__ q16,
                                 _Float16* __restrict__ k16,
                                 _Float16* __restrict__ v16,
                                 _Float16* __restrict__ g16) {
  __shared__ _Float16 stage[8][16][136];   // per-wave 16x128 strip, padded pitch
  const int wave = threadIdx.x >> 5, lane = threadIdx.x & 31;
  const int strip = blockIdx.x * 8 + wave;       // 4096 strips
  const int g = lane >> 4, ln = lane & 15;
  const int row = strip * 16 + ln;

  v16h a[4];
  const _Float16* hrow = h16 + (size_t)row * D_DIM;
#pragma unroll
  for (int c = 0; c < 4; ++c) {
#pragma unroll
    for (int v = 0; v < 8; ++v) {
      const v2h pr = *(const v2h*)(hrow + c * 32 + a_kmap(g, v));
      a[c][2 * v] = pr[0];
      a[c][2 * v + 1] = pr[1];
    }
  }

  const int lr = lane >> 1;              // flush: row handled by this lane
  const int cb = (lane & 1) * 64;        // flush: column base (64 halves each)

  _Float16* dsts[4] = {q16, k16, v16, g16};
  for (int m = 0; m < 4; ++m) {
    _Float16* dst = dsts[m];
#pragma unroll
    for (int j = 0; j < 8; ++j) {
      v8f acc = {};
#pragma unroll
      for (int c = 0; c < 4; ++c) {
        const v16h b =
            *(const v16h*)(wfrag + ((((size_t)m * 8 + j) * 4 + c) * 32 + lane) * 16);
        acc = __builtin_amdgcn_wmma_f32_16x16x32_f16(false, a[c], false, b,
                                                     (short)0, acc, false, false);
      }
      if (m == 3) {  // gate: fused sigmoid
#pragma unroll
        for (int r = 0; r < 8; ++r) acc[r] = 1.0f / (1.0f + __expf(-acc[r]));
      }
#pragma unroll
      for (int r = 0; r < 8; ++r)
        stage[wave][r + 8 * g][j * 16 + ln] = (_Float16)acc[r];
    }
    // coalesced flush: each lane streams 64 contiguous halves of one row
    _Float16* drow = dst + (size_t)(strip * 16 + lr) * D_DIM + cb;
#pragma unroll
    for (int u = 0; u < 8; ++u)
      *(v8h*)(drow + u * 8) = *(const v8h*)(&stage[wave][lr][cb + u * 8]);
  }
}

// ------------------------- kernel 4: attention ------------------------------
// grid (S, H): one workgroup per (row b, head h); 128 threads = 4 waves;
// each wave handles 4 query M-tiles of 16 (whole 256-wide score row in regs).
__global__ void attn_kernel(const _Float16* __restrict__ q16,
                            const _Float16* __restrict__ k16,
                            const _Float16* __restrict__ v16,
                            const float* __restrict__ bias,
                            _Float16* __restrict__ o16) {
  const int b = blockIdx.x, h = blockIdx.y;
  const int wave = threadIdx.x >> 5, lane = threadIdx.x & 31;
  const int g = lane >> 4, ln = lane & 15;

  extern __shared__ __align__(128) char smem[];
  _Float16* vt = (_Float16*)smem;                 // V^T: [32 d][256 jk]
  _Float16* attn = vt + 32 * 256;                 // [4 waves][16][264]
  _Float16* ostage = attn + 4 * 16 * 264;         // [4 waves][16][32]

  // cooperative V^T staging (V tile is shared by all waves of the workgroup)
  for (int jk = threadIdx.x; jk < 256; jk += 128) {
    const _Float16* vrow = v16 + ((size_t)(b * 256 + jk)) * D_DIM + h * HD_DIM;
#pragma unroll
    for (int d = 0; d < 32; ++d) vt[d * 256 + jk] = vrow[d];
  }
  __syncthreads();

  _Float16* myattn = attn + (size_t)wave * 16 * 264;
  _Float16* myost = ostage + (size_t)wave * 16 * 32;
  const float* bias_h = bias + (size_t)h * NPOS;

  for (int mt = 0; mt < 4; ++mt) {
    const int q0 = (wave * 4 + mt) * 16;

    // Q A-fragment (16 x 32, single K step since hd == 32)
    v16h aq;
    const _Float16* qrow = q16 + ((size_t)(b * 256 + q0 + ln)) * D_DIM + h * HD_DIM;
#pragma unroll
    for (int v = 0; v < 8; ++v) {
      const v2h pr = *(const v2h*)(qrow + a_kmap(g, v));
      aq[2 * v] = pr[0];
      aq[2 * v + 1] = pr[1];
    }

    // scores: 16 tiles of 16x16, one WMMA each (K = hd = 32)
    v8f sc[16];
#pragma unroll
    for (int j = 0; j < 16; ++j) {
      const _Float16* kp =
          k16 + ((size_t)(b * 256 + j * 16 + ln)) * D_DIM + h * HD_DIM + 16 * g;
      if (j < 15)
        __builtin_prefetch(kp + 16 * (size_t)D_DIM, 0, 0);  // global_prefetch_b8
      const v16h bk = *(const v16h*)kp;
      v8f z = {};
      sc[j] = __builtin_amdgcn_wmma_f32_16x16x32_f16(false, aq, false, bk,
                                                     (short)0, z, false, false);
    }

    // bias + scale + softmax (row fully resident: 16 tiles x VGPR r per lane)
    float rinv[8];
#pragma unroll
    for (int r = 0; r < 8; ++r) {
      const int qr = q0 + r + 8 * g;
      const float* brow = bias_h + (size_t)qr * 256 + ln;
      float mx = -1e30f;
#pragma unroll
      for (int j = 0; j < 16; ++j) {
        const float t = sc[j][r] * SCALE + brow[j * 16];
        sc[j][r] = t;
        mx = fmaxf(mx, t);
      }
#pragma unroll
      for (int msk = 8; msk >= 1; msk >>= 1) mx = fmaxf(mx, __shfl_xor(mx, msk, 32));
      float sum = 0.0f;
#pragma unroll
      for (int j = 0; j < 16; ++j) {
        const float p = __expf(sc[j][r] - mx);
        sum += p;
        myattn[(r + 8 * g) * 264 + j * 16 + ln] = (_Float16)p;   // ds_store_b16
      }
#pragma unroll
      for (int msk = 8; msk >= 1; msk >>= 1) sum += __shfl_xor(sum, msk, 32);
      rinv[r] = 1.0f / sum;
    }

    // O = P @ V : both N-tiles share each A-fragment (one LDS read per chunk)
    v8f oacc0 = {}, oacc1 = {};
#pragma unroll
    for (int c = 0; c < 8; ++c) {
      v16h ap;
#pragma unroll
      for (int v = 0; v < 8; ++v) {
        const v2h pr = *(const v2h*)(myattn + ln * 264 + c * 32 + a_kmap(g, v));
        ap[2 * v] = pr[0];
        ap[2 * v + 1] = pr[1];
      }
      const v16h bv0 = *(const v16h*)(vt + ln * 256 + c * 32 + 16 * g);
      const v16h bv1 = *(const v16h*)(vt + (16 + ln) * 256 + c * 32 + 16 * g);
      oacc0 = __builtin_amdgcn_wmma_f32_16x16x32_f16(false, ap, false, bv0,
                                                     (short)0, oacc0, false, false);
      oacc1 = __builtin_amdgcn_wmma_f32_16x16x32_f16(false, ap, false, bv1,
                                                     (short)0, oacc1, false, false);
    }
    // scale by 1/rowsum (lane-aligned) and stage, then coalesced flush
#pragma unroll
    for (int r = 0; r < 8; ++r) {
      myost[(r + 8 * g) * 32 + ln] = (_Float16)(oacc0[r] * rinv[r]);
      myost[(r + 8 * g) * 32 + 16 + ln] = (_Float16)(oacc1[r] * rinv[r]);
    }
    _Float16* orow =
        o16 + ((size_t)(b * 256 + q0 + ln)) * D_DIM + h * HD_DIM + g * 16;
    const _Float16* srow = myost + ln * 32 + g * 16;
    *(v8h*)(orow + 0) = *(const v8h*)(srow + 0);
    *(v8h*)(orow + 8) = *(const v8h*)(srow + 8);
  }
}

// ------------------------- kernel 5: gate * o, then @ w_o^T -----------------
// 128 threads = 4 waves; fp32 C tiles staged in LDS, flushed as b128 stores.
__global__ void gate_out_kernel(const _Float16* __restrict__ o16,
                                const _Float16* __restrict__ g16,
                                const _Float16* __restrict__ wfrag,
                                float* __restrict__ out) {
  __shared__ float stage[4][16][132];
  const int wave = threadIdx.x >> 5, lane = threadIdx.x & 31;
  const int strip = blockIdx.x * 4 + wave;
  const int g = lane >> 4, ln = lane & 15;
  const int row = strip * 16 + ln;

  v16h a[4];
  const _Float16* orow = o16 + (size_t)row * D_DIM;
  const _Float16* grow = g16 + (size_t)row * D_DIM;
#pragma unroll
  for (int c = 0; c < 4; ++c) {
#pragma unroll
    for (int v = 0; v < 8; ++v) {
      const int K = a_kmap(g, v);
      const v2h po = *(const v2h*)(orow + c * 32 + K);
      const v2h pg = *(const v2h*)(grow + c * 32 + K);
      a[c][2 * v] = (_Float16)(po[0] * pg[0]);
      a[c][2 * v + 1] = (_Float16)(po[1] * pg[1]);
    }
  }

  const _Float16* wofrag = wfrag + (size_t)4 * 16384;   // matrix index 4 = w_o
#pragma unroll
  for (int j = 0; j < 8; ++j) {
    v8f acc = {};
#pragma unroll
    for (int c = 0; c < 4; ++c) {
      const v16h b = *(const v16h*)(wofrag + (((size_t)j * 4 + c) * 32 + lane) * 16);
      acc = __builtin_amdgcn_wmma_f32_16x16x32_f16(false, a[c], false, b,
                                                   (short)0, acc, false, false);
    }
#pragma unroll
    for (int r = 0; r < 8; ++r)
      stage[wave][r + 8 * g][j * 16 + ln] = acc[r];
  }
  // coalesced flush: each lane streams 64 contiguous floats of one row
  const int lr = lane >> 1, cb = (lane & 1) * 64;
  float* drow = out + (size_t)(strip * 16 + lr) * D_DIM + cb;
#pragma unroll
  for (int u = 0; u < 16; ++u)
    *(float4*)(drow + u * 4) = *(const float4*)(&stage[wave][lr][cb + u * 4]);
}

// ------------------------- launch -------------------------------------------
extern "C" void kernel_launch(void* const* d_in, const int* in_sizes, int n_in,
                              void* d_out, int out_size, void* d_ws, size_t ws_size,
                              hipStream_t stream) {
  const float* x      = (const float*)d_in[0];
  const float* ln_w   = (const float*)d_in[1];
  const float* ln_b   = (const float*)d_in[2];
  const float* w_bias = (const float*)d_in[3];
  const float* w_q    = (const float*)d_in[4];
  const float* w_k    = (const float*)d_in[5];
  const float* w_v    = (const float*)d_in[6];
  const float* w_g    = (const float*)d_in[7];
  const float* w_o    = (const float*)d_in[8];
  float* out = (float*)d_out;

  // workspace carve-up (all offsets 256B-aligned)
  const size_t HB = (size_t)NPOS * D_DIM;      // elements per fp16 plane
  char* ws = (char*)d_ws;
  _Float16* h16 = (_Float16*)ws;
  _Float16* q16 = h16 + HB;
  _Float16* k16 = q16 + HB;
  _Float16* v16 = k16 + HB;
  _Float16* g16 = v16 + HB;
  _Float16* o16 = g16 + HB;
  float*    bias  = (float*)(o16 + HB);                 // H*S*S f32 = 1 MB
  _Float16* wfrag = (_Float16*)(bias + (size_t)H_DIM * NPOS);  // 5*16384 f16

  // 1. layernorm + per-head triangle bias
  ln_bias_kernel<<<NPOS / 8, 256, 0, stream>>>(x, ln_w, ln_b, w_bias, h16, bias);

  // 2. weight repack into WMMA B-fragment layout
  prepack_kernel<<<5, 256, 0, stream>>>(w_q, w_k, w_v, w_g, w_o, wfrag);

  // 3. q/k/v/g projections (sigmoid fused into g)
  proj_gemm_kernel<<<(NPOS / 16) / 8, 256, 0, stream>>>(h16, wfrag, q16, k16, v16, g16);

  // 4. attention per (row b, head h)
  const size_t attn_lds =
      (size_t)(32 * 256 + 4 * 16 * 264 + 4 * 16 * 32) * sizeof(_Float16); // 54272 B
  attn_kernel<<<dim3(S_DIM, H_DIM), 128, attn_lds, stream>>>(q16, k16, v16, bias, o16);

  // 5. gate + output projection (fp32 out)
  gate_out_kernel<<<(NPOS / 16) / 4, 128, 0, stream>>>(o16, g16, wfrag, out);
}
